// Cobra_7447473291763
// MI455X (gfx1250) — matrix operand order repaired and checked by reading, compiled, not run on hardware
//
#include <hip/hip_runtime.h>
#include <hip/hip_bf16.h>

// ---------------- model constants ----------------
#define B_      4
#define T_      1024
#define DM      768          // D_MODEL
#define DI      1536         // D_INNER
#define DSTATE  64
#define NH      24
#define HD      64
#define DZ      3224         // D_IN_PROJ = 2*DI + 2*DSTATE + NH
#define DZP     3264         // DZ padded to multiple of 64 (GEMM tile-exact)
#define CD      1664         // CONV_DIM = DI + 2*DSTATE
#define MTOT    4096         // B_*T_
#define HATT    8
#define EATT    96
#define CDIM    2048
#define EPS_    1e-5f

typedef __attribute__((ext_vector_type(16))) __bf16 v16bf;
typedef __attribute__((ext_vector_type(8)))  float  v8f;

// ---------------- async global->LDS staging (gfx1250 ASYNCcnt path) ----------------
#if __has_builtin(__builtin_amdgcn_global_load_async_to_lds_b128)
#define HAVE_ASYNC_LDS 1
typedef int a_v4i __attribute__((vector_size(16)));
typedef __attribute__((address_space(1))) a_v4i ga_v4i;   // global v4i
typedef __attribute__((address_space(3))) a_v4i la_v4i;   // LDS v4i
__device__ __forceinline__ void async_g2l_b128(const void* g, void* l) {
    // AS3 offset = low 32 bits of generic LDS pointer
    __builtin_amdgcn_global_load_async_to_lds_b128(
        (ga_v4i*)(unsigned long long)(size_t)g,
        (la_v4i*)(unsigned int)(size_t)l, 0, 0);
}
__device__ __forceinline__ void wait_async_lds() {
#if __has_builtin(__builtin_amdgcn_s_wait_asynccnt)
    __builtin_amdgcn_s_wait_asynccnt(0);
#else
    asm volatile("s_wait_asynccnt 0x0" ::: "memory");
#endif
}
#else
#define HAVE_ASYNC_LDS 0
#endif

// ---------------- small device helpers ----------------
__device__ __forceinline__ float sigmoid_(float x) { return 1.0f / (1.0f + expf(-x)); }
__device__ __forceinline__ float silu_(float x)    { return x * sigmoid_(x); }

__device__ __forceinline__ unsigned short bf16_rne_(float f) {
    unsigned int u = __float_as_uint(f);
    u += 0x7FFFu + ((u >> 16) & 1u);
    return (unsigned short)(u >> 16);
}

__device__ __forceinline__ float block_reduce_sum_256(float v, float* sm) {
    int tid = threadIdx.x;
    sm[tid] = v; __syncthreads();
    for (int off = 128; off > 0; off >>= 1) {
        if (tid < off) sm[tid] += sm[tid + off];
        __syncthreads();
    }
    float r = sm[0]; __syncthreads();
    return r;
}

// ---------------- f32 -> bf16 (RNE), flat ----------------
__global__ void cvt_bf16_kernel(const float* __restrict__ src,
                                unsigned short* __restrict__ dst, long n) {
    long i = (long)blockIdx.x * 256 + threadIdx.x;
    if (i < n) dst[i] = bf16_rne_(src[i]);
}

// ---- f32 W[K,N] -> bf16 Wt[NP,K] transposed + zero-padded (one-time per GEMM) ----
__global__ void cvt_wt_bf16_kernel(const float* __restrict__ src,
                                   unsigned short* __restrict__ dst,
                                   int K, int N, long total /* = NP*K */) {
    long i = (long)blockIdx.x * 256 + threadIdx.x;
    if (i < total) {
        int  k = (int)(i % K);
        long n = i / K;
        dst[i] = (n < N) ? bf16_rne_(src[(long)k * N + n]) : (unsigned short)0;
    }
}

// ---------------- LayerNorm (one block per row) ----------------
__global__ __launch_bounds__(256)
void ln_kernel(const float* __restrict__ x, const float* __restrict__ g,
               const float* __restrict__ b, float* __restrict__ out, int D) {
    __shared__ float sm[256];
    long row = blockIdx.x;
    int tid = threadIdx.x;
    const float* xr = x + row * D;
    float s = 0.f, ss = 0.f;
    for (int c = tid; c < D; c += 256) { float v = xr[c]; s += v; ss += v * v; }
    float tot  = block_reduce_sum_256(s,  sm);
    float tot2 = block_reduce_sum_256(ss, sm);
    float mean = tot / (float)D;
    float var  = tot2 / (float)D - mean * mean;
    float inv  = rsqrtf(var + EPS_);
    float* orow = out + row * D;
    for (int c = tid; c < D; c += 256) orow[c] = (xr[c] - mean) * inv * g[c] + b[c];
}

// ---------------- WMMA GEMM: out = act(A@Wt^T + bias) + resid ----------------
// A:  [M,K]  bf16 row-major.  Wt: [N,K] bf16 (weights pre-transposed, N padded).
// Tile-exact: M%64==0, N%64==0, K%64==0 -> no guards anywhere.
// Block = 256 thr = 8 waves; 64x64 block tile; wave owns 16x32 (2 accs).
// K staged in 64-chunks, LDS double-buffered: 1 barrier + 4 WMMA per stage.
// Staging uses GLOBAL_LOAD_ASYNC_TO_LDS_B128 (ASYNCcnt) when available.
__global__ __launch_bounds__(256)
void gemm_wmma_kernel(const unsigned short* __restrict__ A,
                      const unsigned short* __restrict__ Wt,
                      const float* __restrict__ bias,     // [N] or null
                      const float* __restrict__ resid,    // [M,N] or null
                      float* __restrict__ out,            // [M,N]
                      int M, int N, int K, int act) {
    __shared__ __attribute__((aligned(16))) unsigned short lA[2][64][72];
    __shared__ __attribute__((aligned(16))) unsigned short lB[2][64][72];

    int tid  = threadIdx.x;
    int lane = tid & 31;
    int wave = tid >> 5;
    int wm = wave >> 1;            // 0..3  (16-row slab)
    int wn = wave & 1;             // 0..1  (32-col slab)
    int gm0 = blockIdx.x * 64;
    int gn0 = blockIdx.y * 64;

    v8f acc0 = {0.f,0.f,0.f,0.f,0.f,0.f,0.f,0.f};
    v8f acc1 = {0.f,0.f,0.f,0.f,0.f,0.f,0.f,0.f};

    // staging map (same for A and B since both are [*,K] row-major):
    // thread -> (row tid>>2, 16 k at (tid&3)*16) = 2x b128 per matrix
    int srow = tid >> 2;
    int scol = (tid & 3) * 16;
    const unsigned short* Ap = A  + (long)(gm0 + srow) * K + scol;
    const unsigned short* Bp = Wt + (long)(gn0 + srow) * K + scol;

    // fragment gather coords (contiguous k order: element s of lane half h is k = 16h + s)
    int rloc = wm * 16 + (lane & 15);
    int c0   = wn * 32 + (lane & 15);
    int c1   = c0 + 16;
    int hiA  = (lane >> 4) * 16;    // A: hi lanes hold K+16 sub-block
    int hiB  = (lane >> 4) * 16;    // B: hi lanes hold K+16 sub-block

    int nk = K >> 6;                // number of 64-wide K chunks

    // ---- stage one 64-K chunk into LDS buffer `buf` ----
    auto stage = [&](int k0, int buf) {
#if HAVE_ASYNC_LDS
        async_g2l_b128(Ap + k0,     &lA[buf][srow][scol]);
        async_g2l_b128(Ap + k0 + 8, &lA[buf][srow][scol + 8]);
        async_g2l_b128(Bp + k0,     &lB[buf][srow][scol]);
        async_g2l_b128(Bp + k0 + 8, &lB[buf][srow][scol + 8]);
#else
        uint4 a0 = *reinterpret_cast<const uint4*>(Ap + k0);
        uint4 a1 = *reinterpret_cast<const uint4*>(Ap + k0 + 8);
        uint4 b0 = *reinterpret_cast<const uint4*>(Bp + k0);
        uint4 b1 = *reinterpret_cast<const uint4*>(Bp + k0 + 8);
        *reinterpret_cast<uint4*>(&lA[buf][srow][scol])     = a0;
        *reinterpret_cast<uint4*>(&lA[buf][srow][scol + 8]) = a1;
        *reinterpret_cast<uint4*>(&lB[buf][srow][scol])     = b0;
        *reinterpret_cast<uint4*>(&lB[buf][srow][scol + 8]) = b1;
#endif
    };

    stage(0, 0);                    // prologue

    for (int i = 0; i < nk; ++i) {
#if HAVE_ASYNC_LDS
        wait_async_lds();           // our async copies into buffer i&1 are done
#endif
        __syncthreads();            // ...and visible to all waves

        // issue next chunk's staging into the other buffer (overlaps the WMMAs)
        if (i + 1 < nk) {
            stage((i + 1) << 6, (i + 1) & 1);
            if (i + 2 < nk) {
                __builtin_prefetch((const void*)(Ap + ((i + 2) << 6)), 0, 1);
                __builtin_prefetch((const void*)(Bp + ((i + 2) << 6)), 0, 1);
            }
        }

        // compute on current buffer: 2 k-chunks x (A-frag reused over 2 B-frags)
        const unsigned short (*cA)[72] = lA[i & 1];
        const unsigned short (*cB)[72] = lB[i & 1];
        #pragma unroll
        for (int kk = 0; kk < 64; kk += 32) {
            union { uint4 q[2]; v16bf v; } Ua, Ub0, Ub1;
            Ua.q[0]  = *reinterpret_cast<const uint4*>(&cA[rloc][kk + hiA]);
            Ua.q[1]  = *reinterpret_cast<const uint4*>(&cA[rloc][kk + hiA + 8]);
            Ub0.q[0] = *reinterpret_cast<const uint4*>(&cB[c0  ][kk + hiB]);
            Ub0.q[1] = *reinterpret_cast<const uint4*>(&cB[c0  ][kk + hiB + 8]);
            Ub1.q[0] = *reinterpret_cast<const uint4*>(&cB[c1  ][kk + hiB]);
            Ub1.q[1] = *reinterpret_cast<const uint4*>(&cB[c1  ][kk + hiB + 8]);
            acc0 = __builtin_amdgcn_wmma_f32_16x16x32_bf16(
                       false, Ua.v, false, Ub0.v, (short)0, acc0, false, false);
            acc1 = __builtin_amdgcn_wmma_f32_16x16x32_bf16(
                       false, Ua.v, false, Ub1.v, (short)0, acc1, false, false);
        }
    }

    // ---- epilogue: C/D layout: n = lane&15 (+tile), m = r + 8*(lane>=16) ----
    int n0 = gn0 + wn * 32 + (lane & 15);
    int n1 = n0 + 16;
    int mbase = gm0 + wm * 16 + ((lane >> 4) << 3);
    float bv0 = bias ? bias[n0] : 0.f;
    float bv1 = bias ? bias[n1] : 0.f;
    #pragma unroll
    for (int r = 0; r < 8; ++r) {
        long m = mbase + r;
        float v0 = acc0[r] + bv0;
        float v1 = acc1[r] + bv1;
        if (act == 1) { v0 = v0 * sigmoid_(v0); v1 = v1 * sigmoid_(v1); }
        if (resid) { v0 += resid[m * N + n0]; v1 += resid[m * N + n1]; }
        out[m * N + n0] = v0;
        out[m * N + n1] = v1;
    }
}

// ---------------- depthwise conv (D_CONV=4, causal) + SiLU ----------------
__global__ __launch_bounds__(256)
void conv_silu_kernel(const float* __restrict__ zxbcdt,    // [MTOT, DZP]
                      const float* __restrict__ cw,        // [CD,4]
                      const float* __restrict__ cb,        // [CD]
                      float* __restrict__ xBC) {
    long bt = blockIdx.x;
    int t = (int)(bt & (T_ - 1));
    for (int c = threadIdx.x; c < CD; c += 256) {
        float acc = cb[c];
        #pragma unroll
        for (int k = 0; k < 4; ++k) {
            int tt = t - 3 + k;
            if (tt >= 0)
                acc += zxbcdt[(bt - 3 + k) * DZP + DI + c] * cw[c * 4 + k];
        }
        xBC[bt * CD + c] = silu_(acc);
    }
}

// ---------------- dt = softplus(dt_raw + bias), dA = exp(-exp(A_log)*dt) ----------------
__global__ void dt_kernel(const float* __restrict__ zxbcdt,
                          const float* __restrict__ dt_bias,
                          const float* __restrict__ A_log,
                          float* __restrict__ dtc, float* __restrict__ dAc, int total) {
    int i = blockIdx.x * 256 + threadIdx.x;
    if (i < total) {
        int h = i % NH;
        long bt = i / NH;
        float x = zxbcdt[bt * DZP + (DI + CD) + h] + dt_bias[h];
        float sp = (x > 20.f) ? x : log1pf(expf(x));
        dtc[i] = sp;
        dAc[i] = expf(-expf(A_log[h]) * sp);
    }
}

// ---------------- selective scan: one block per (b, head) ----------------
__global__ __launch_bounds__(256)
void scan_kernel(const float* __restrict__ xBC,
                 const float* __restrict__ dtc, const float* __restrict__ dAc,
                 const float* __restrict__ Dp, float* __restrict__ y) {
    int b  = blockIdx.x / NH;
    int hh = blockIdx.x % NH;
    int tid = threadIdx.x;
    int p  = tid >> 2;            // 0..63 (head dim)
    int n0 = (tid & 3) * 16;      // 16-state slice
    __shared__ float sB[64], sC[64], sX[64];
    float s[16];
    #pragma unroll
    for (int j = 0; j < 16; ++j) s[j] = 0.f;
    float Dh = Dp[hh];

    for (int t = 0; t < T_; ++t) {
        long bt = (long)b * T_ + t;
        const float* xrow = xBC + bt * CD;
        if (tid < 64)        sB[tid]       = xrow[DI + tid];
        else if (tid < 128)  sC[tid - 64]  = xrow[DI + DSTATE + (tid - 64)];
        else if (tid < 192)  sX[tid - 128] = xrow[hh * HD + (tid - 128)];
        __syncthreads();

        float dA_t = dAc[bt * NH + hh];
        float dt_t = dtc[bt * NH + hh];
        float xh   = sX[p];
        float dtx  = dt_t * xh;
        float part = 0.f;
        #pragma unroll
        for (int j = 0; j < 16; ++j) {
            s[j] = dA_t * s[j] + dtx * sB[n0 + j];
            part += s[j] * sC[n0 + j];
        }
        part += __shfl_xor(part, 1, 32);
        part += __shfl_xor(part, 2, 32);
        if ((tid & 3) == 0)
            y[bt * DI + hh * HD + p] = part + Dh * xh;
        __syncthreads();
    }
}

// ---------------- y = RMSNorm(y * silu(z)) * w  (in place) ----------------
__global__ __launch_bounds__(256)
void gate_rms_kernel(float* __restrict__ y, const float* __restrict__ zxbcdt,
                     const float* __restrict__ w) {
    __shared__ float sm[256];
    long row = blockIdx.x;
    int tid = threadIdx.x;
    float v[6]; float ss = 0.f;
    #pragma unroll
    for (int i = 0; i < 6; ++i) {
        int c = tid + i * 256;
        float yv = y[row * DI + c];
        float zv = zxbcdt[row * DZP + c];
        float pv = yv * silu_(zv);
        v[i] = pv; ss += pv * pv;
    }
    float tot = block_reduce_sum_256(ss, sm);
    float inv = rsqrtf(tot / (float)DI + EPS_);
    #pragma unroll
    for (int i = 0; i < 6; ++i) {
        int c = tid + i * 256;
        y[row * DI + c] = v[i] * inv * w[c];
    }
}

// ---------------- attention scores: raw.mean(heads) per (b,t) ----------------
__global__ __launch_bounds__(256)
void attn_kernel(const float* __restrict__ h, const float* __restrict__ wa,
                 const float* __restrict__ ba, const float* __restrict__ wb,
                 const float* __restrict__ bb, const float* __restrict__ wc,
                 const float* __restrict__ bc, float* __restrict__ rawm) {
    __shared__ float hrow[DM];
    __shared__ float red[256];
    long bt = blockIdx.x;
    int tid = threadIdx.x;
    for (int c = tid; c < DM; c += 256) hrow[c] = h[bt * DM + c];
    __syncthreads();

    float part = 0.f;
    int hh = tid & 7;                 // 256 % 8 == 0 -> each thread fixed head
    for (int o = tid; o < EATT * HATT; o += 256) {
        int f = o >> 3;
        float da = ba[hh * EATT + f];
        float dg = bb[hh * EATT + f];
        for (int e = 0; e < EATT; ++e) {
            float hv = hrow[e * HATT + hh];
            da += hv * wa[(hh * EATT + e) * EATT + f];
            dg += hv * wb[(hh * EATT + e) * EATT + f];
        }
        part += tanhf(da) * sigmoid_(dg) * wc[hh * EATT + f];
    }
    red[tid] = part; __syncthreads();
    for (int off = 128; off >= 8; off >>= 1) {  // offsets multiples of 8 keep head groups
        if (tid < off) red[tid] += red[tid + off];
        __syncthreads();
    }
    if (tid == 0) {
        float s = 0.f;
        for (int q = 0; q < HATT; ++q) s += red[q] + bc[q];
        rawm[bt] = s / (float)HATT;
    }
}

// ---------------- softmax over T per batch ----------------
__global__ __launch_bounds__(256)
void softmax_kernel(const float* __restrict__ rawm, float* __restrict__ attw) {
    __shared__ float sm[256];
    int b = blockIdx.x, tid = threadIdx.x;
    const float* r = rawm + (long)b * T_;
    float mx = -1e30f;
    for (int t = tid; t < T_; t += 256) mx = fmaxf(mx, r[t]);
    sm[tid] = mx; __syncthreads();
    for (int off = 128; off > 0; off >>= 1) {
        if (tid < off) sm[tid] = fmaxf(sm[tid], sm[tid + off]);
        __syncthreads();
    }
    mx = sm[0]; __syncthreads();
    float s = 0.f;
    for (int t = tid; t < T_; t += 256) s += expf(r[t] - mx);
    float tot = block_reduce_sum_256(s, sm);
    for (int t = tid; t < T_; t += 256)
        attw[(long)b * T_ + t] = expf(r[t] - mx) / tot;
}

// ---------------- pooled[b,d] = sum_t attw[b,t]*x[b,t,d] ----------------
__global__ __launch_bounds__(256)
void pool_kernel(const float* __restrict__ attw, const float* __restrict__ x,
                 float* __restrict__ pooled) {
    int b = blockIdx.x, tid = threadIdx.x;
    for (int d = tid; d < DM; d += 256) {
        float acc = 0.f;
        for (int t = 0; t < T_; ++t)
            acc += attw[(long)b * T_ + t] * x[((long)b * T_ + t) * DM + d];
        pooled[(long)b * DM + d] = acc;
    }
}

// ---------------- tiny head matvec (M=4 rows) ----------------
__global__ void matvec_kernel(const float* __restrict__ A, const float* __restrict__ W,
                              const float* __restrict__ bias, float* __restrict__ out,
                              int R, int D, int N, int act) {
    int i = blockIdx.x * 256 + threadIdx.x;
    if (i >= R * N) return;
    int r = i / N, j = i % N;
    float acc = bias[j];
    for (int d = 0; d < D; ++d) acc += A[(long)r * D + d] * W[(long)d * N + j];
    if (act) acc = acc * sigmoid_(acc);
    out[i] = acc;
}

// ---------------- batchnorm (inference) ----------------
__global__ void bn_kernel(const float* __restrict__ z, const float* __restrict__ g,
                          const float* __restrict__ b, const float* __restrict__ mean,
                          const float* __restrict__ var, float* __restrict__ out,
                          int total, int C) {
    int i = blockIdx.x * 256 + threadIdx.x;
    if (i < total) {
        int c = i % C;
        out[i] = (z[i] - mean[c]) * rsqrtf(var[c] + EPS_) * g[c] + b[c];
    }
}

// ============================================================
extern "C" void kernel_launch(void* const* d_in, const int* in_sizes, int n_in,
                              void* d_out, int out_size, void* d_ws, size_t ws_size,
                              hipStream_t stream) {
    (void)in_sizes; (void)n_in; (void)out_size; (void)ws_size;

    const float* x         = (const float*)d_in[0];
    const float* emb_ln_g  = (const float*)d_in[1];
    const float* emb_ln_b  = (const float*)d_in[2];
    const float* emb_w1    = (const float*)d_in[3];
    const float* emb_b1    = (const float*)d_in[4];
    const float* emb_w2    = (const float*)d_in[5];
    const float* emb_b2    = (const float*)d_in[6];
    const float* mam_ln_g  = (const float*)d_in[7];
    const float* mam_ln_b  = (const float*)d_in[8];
    const float* mam_in_w  = (const float*)d_in[9];
    const float* mam_conv_w= (const float*)d_in[10];
    const float* mam_conv_b= (const float*)d_in[11];
    const float* mam_dt_b  = (const float*)d_in[12];
    const float* mam_A_log = (const float*)d_in[13];
    const float* mam_D     = (const float*)d_in[14];
    const float* mam_rms_w = (const float*)d_in[15];
    const float* mam_out_w = (const float*)d_in[16];
    const float* enc_ln_g  = (const float*)d_in[17];
    const float* enc_ln_b  = (const float*)d_in[18];
    const float* cob_ln_g  = (const float*)d_in[19];
    const float* cob_ln_b  = (const float*)d_in[20];
    const float* attn_wa   = (const float*)d_in[21];
    const float* attn_ba   = (const float*)d_in[22];
    const float* attn_wb   = (const float*)d_in[23];
    const float* attn_bb   = (const float*)d_in[24];
    const float* attn_wc   = (const float*)d_in[25];
    const float* attn_bc   = (const float*)d_in[26];
    const float* proj_ln_g = (const float*)d_in[27];
    const float* proj_ln_b = (const float*)d_in[28];
    const float* proj_w1   = (const float*)d_in[29];
    const float* proj_b1   = (const float*)d_in[30];
    const float* proj_w2   = (const float*)d_in[31];
    const float* proj_b2   = (const float*)d_in[32];
    const float* bn_g      = (const float*)d_in[33];
    const float* bn_b      = (const float*)d_in[34];
    const float* bn_mean   = (const float*)d_in[35];
    const float* bn_var    = (const float*)d_in[36];

    float* out = (float*)d_out;          // [0,8192): feats; [8192,11264): pooled

    // ---- workspace carve-out ----
    char* wsp = (char*)d_ws;
    auto alloc = [&](size_t bytes) -> void* {
        void* p = (void*)wsp;
        wsp += (bytes + 255) & ~(size_t)255;
        return p;
    };
    float* h      = (float*)alloc((size_t)MTOT * DM * 4);
    float* tmp1   = (float*)alloc((size_t)MTOT * DM * 4);
    float* tmp2   = (float*)alloc((size_t)MTOT * DM * 4);
    float* zxbcdt = (float*)alloc((size_t)MTOT * DZP * 4);   // padded stride
    float* xBC    = (float*)alloc((size_t)MTOT * CD * 4);
    float* ybuf   = (float*)alloc((size_t)MTOT * DI * 4);
    float* dtc    = (float*)alloc((size_t)MTOT * NH * 4);
    float* dAc    = (float*)alloc((size_t)MTOT * NH * 4);
    float* rawm   = (float*)alloc((size_t)MTOT * 4);
    float* attw   = (float*)alloc((size_t)MTOT * 4);
    float* pln    = (float*)alloc((size_t)B_ * DM * 4);
    float* ph     = (float*)alloc((size_t)B_ * 4 * DM * 4);
    float* zf     = (float*)alloc((size_t)B_ * CDIM * 4);
    unsigned short* abf = (unsigned short*)alloc((size_t)MTOT * DI * 2);
    unsigned short* wbf = (unsigned short*)alloc((size_t)DZP * DM * 2);  // [NP,K] transposed

    auto cvt = [&](const float* src, unsigned short* dst, long n) {
        cvt_bf16_kernel<<<(unsigned)((n + 255) / 256), 256, 0, stream>>>(src, dst, n);
    };
    auto cvtw = [&](const float* src, unsigned short* dst, int K, int N, int NP) {
        long total = (long)NP * K;
        cvt_wt_bf16_kernel<<<(unsigned)((total + 255) / 256), 256, 0, stream>>>(
            src, dst, K, N, total);
    };
    auto gemm = [&](const unsigned short* A, const unsigned short* Wt, const float* bias,
                    const float* resid, float* o, int M, int N, int K, int act) {
        dim3 grid((unsigned)(M / 64), (unsigned)(N / 64));
        gemm_wmma_kernel<<<grid, 256, 0, stream>>>(A, Wt, bias, resid, o, M, N, K, act);
    };

    // ---- embedding MLP ----
    ln_kernel<<<MTOT, 256, 0, stream>>>(x, emb_ln_g, emb_ln_b, tmp1, DM);
    cvt(tmp1, abf, (long)MTOT * DM);
    cvtw(emb_w1, wbf, DM, DM, DM);
    gemm(abf, wbf, emb_b1, nullptr, tmp2, MTOT, DM, DM, 1);       // silu
    cvt(tmp2, abf, (long)MTOT * DM);
    cvtw(emb_w2, wbf, DM, DM, DM);
    gemm(abf, wbf, emb_b2, nullptr, h, MTOT, DM, DM, 0);

    // ---- 4 Mamba2 layers ----
    for (int l = 0; l < 4; ++l) {
        ln_kernel<<<MTOT, 256, 0, stream>>>(h, mam_ln_g + l * DM, mam_ln_b + l * DM, tmp1, DM);
        cvt(tmp1, abf, (long)MTOT * DM);
        cvtw(mam_in_w + (long)l * DM * DZ, wbf, DM, DZ, DZP);     // transpose + pad 3224->3264
        gemm(abf, wbf, nullptr, nullptr, zxbcdt, MTOT, DZP, DM, 0);

        conv_silu_kernel<<<MTOT, 256, 0, stream>>>(zxbcdt, mam_conv_w + (long)l * CD * 4,
                                                   mam_conv_b + (long)l * CD, xBC);
        dt_kernel<<<(MTOT * NH + 255) / 256, 256, 0, stream>>>(
            zxbcdt, mam_dt_b + l * NH, mam_A_log + l * NH, dtc, dAc, MTOT * NH);
        scan_kernel<<<B_ * NH, 256, 0, stream>>>(xBC, dtc, dAc, mam_D + l * NH, ybuf);
        gate_rms_kernel<<<MTOT, 256, 0, stream>>>(ybuf, zxbcdt, mam_rms_w + (long)l * DI);

        cvt(ybuf, abf, (long)MTOT * DI);
        cvtw(mam_out_w + (long)l * DI * DM, wbf, DI, DM, DM);
        gemm(abf, wbf, nullptr, h /*residual*/, h, MTOT, DM, DI, 0);
    }

    // ---- final norms + attention pooling ----
    ln_kernel<<<MTOT, 256, 0, stream>>>(h, enc_ln_g, enc_ln_b, tmp1, DM);
    ln_kernel<<<MTOT, 256, 0, stream>>>(tmp1, cob_ln_g, cob_ln_b, tmp2, DM);
    attn_kernel<<<MTOT, 256, 0, stream>>>(tmp2, attn_wa, attn_ba, attn_wb, attn_bb,
                                          attn_wc, attn_bc, rawm);
    softmax_kernel<<<B_, 256, 0, stream>>>(rawm, attw);
    pool_kernel<<<B_, 256, 0, stream>>>(attw, x, out + B_ * CDIM);   // pooled -> output tail

    // ---- projection head (M=4: plain matvecs) ----
    ln_kernel<<<B_, 256, 0, stream>>>(out + B_ * CDIM, proj_ln_g, proj_ln_b, pln, DM);
    matvec_kernel<<<(B_ * 4 * DM + 255) / 256, 256, 0, stream>>>(
        pln, proj_w1, proj_b1, ph, B_, DM, 4 * DM, 1);               // silu
    matvec_kernel<<<(B_ * CDIM + 255) / 256, 256, 0, stream>>>(
        ph, proj_w2, proj_b2, zf, B_, 4 * DM, CDIM, 0);
    bn_kernel<<<(B_ * CDIM + 255) / 256, 256, 0, stream>>>(
        zf, bn_g, bn_b, bn_mean, bn_var, out, B_ * CDIM, CDIM);
}
